// Encoder_62715112456845
// MI455X (gfx1250) — matrix operand order repaired and checked by reading, not compiled, not run on hardware
//
#include <hip/hip_runtime.h>
#include <hip/hip_bf16.h>
#include <math.h>

// ---------------------------------------------------------------------------
// FISTA sparse coding, MI455X (gfx1250, wave32, WMMA).
// Persistent-column design: each wave owns 16 of the 40960 independent
// columns and runs all 40 FISTA iterations with state resident in LDS.
// A[161x161] @ y[161x16] via split-f16 WMMA (3 products ~ fp32 accuracy).
// All LDS state kept in WMMA fragment layouts -> pure b128 DS traffic.
// ---------------------------------------------------------------------------

typedef _Float16 v16h __attribute__((ext_vector_type(16)));
typedef _Float16 v8h  __attribute__((ext_vector_type(8)));
typedef float    v8f  __attribute__((ext_vector_type(8)));

#define BB      2
#define TT      10
#define NPOLES  40
#define PP      20480
#define KD      161          // 4*NPOLES + 1
#define MAXIT   40
#define GAMMAF  0.01f

#define NI      11           // row tiles: 176 / 16
#define NK      6            // k tiles:   192 / 32
#define FRAG_HALVES (2*NI*NK*512)     // A hi+lo splits, 512 halves / 16x32 tile
#define FRAGB   (FRAG_HALVES*2)       // 135168 bytes
#define WAVES   4
#define COLS_PER_WG (WAVES*16)

// per-wave LDS state
#define YFRAG_HALVES (2*NK*512)       // y as split-f16 B fragments: 6144 halves
#define DT_FLOATS (NI*2*16*8)         // C/D-tile layout: 2816 floats
#define WAVE_STATE_BYTES (YFRAG_HALVES*2 + 2*DT_FLOATS*4)   // 34816 B

// d_ws layout (bytes)
#define WS_D    64
#define WS_FRAG 8192

// ---------------------------------------------------------------------------
// Kernel 1: build dictionary, A = I - DtD/L (Frobenius L), emit WMMA-ready
// split-f16 A fragments (16x32 A tiles, contiguous k = 16*half + e per lane, same order as B)
// ---------------------------------------------------------------------------
__global__ void __launch_bounds__(256) prep_kernel(const float* __restrict__ Drr,
                                                   const float* __restrict__ Dth,
                                                   float* __restrict__ hdr,
                                                   float* __restrict__ wsD,
                                                   _Float16* __restrict__ wsFrag) {
  __shared__ float Dl[KD * TT];    // column-major: Dl[k*TT + t]
  __shared__ float red[256];
  __shared__ float s_linv;
  const int tid = threadIdx.x;

  if (tid < KD) {
    float vals[TT];
    if (tid == 0) {
      #pragma unroll
      for (int t = 0; t < TT; ++t) vals[t] = 1.f;
    } else {
      const int idx = tid - 1, g = idx / NPOLES, j = idx % NPOLES;
      const float r = Drr[j], th = Dth[j];
      float pr = 1.f;
      for (int t = 0; t < TT; ++t) {
        const float ang = th * (float)t;
        const float base = (g < 2) ? cosf(ang) : sinf(ang);
        const float sgn = (((g == 1) || (g == 3)) && (t & 1)) ? -1.f : 1.f;
        vals[t] = sgn * pr * base;
        pr *= r;
      }
    }
    float nrm = 0.f;
    #pragma unroll
    for (int t = 0; t < TT; ++t) nrm += vals[t] * vals[t];
    nrm = sqrtf(nrm);
    if (nrm == 0.f) nrm = sqrtf((float)TT);
    for (int t = 0; t < TT; ++t) {
      const float v = vals[t] / nrm;
      Dl[tid * TT + t] = v;
      wsD[tid * TT + t] = v;
    }
  }
  __syncthreads();

  // Frobenius norm of DtD (recompute dot products; reduction length only 10)
  float acc = 0.f;
  for (int i = tid; i < KD * KD; i += 256) {
    const int m = i / KD, n = i % KD;
    float d = 0.f;
    #pragma unroll
    for (int t = 0; t < TT; ++t) d += Dl[m * TT + t] * Dl[n * TT + t];
    acc += d * d;
  }
  red[tid] = acc;
  __syncthreads();
  for (int s = 128; s > 0; s >>= 1) {
    if (tid < s) red[tid] += red[tid + s];
    __syncthreads();
  }
  if (tid == 0) {
    const float L = sqrtf(red[0]);
    const float linv = 1.f / L;
    s_linv = linv;
    hdr[0] = linv;
    hdr[1] = GAMMAF * linv;
  }
  __syncthreads();
  const float linv = s_linv;

  // Emit split-f16 A fragments: [split][tile=i*NK+kt][lane][e]
  for (int idx = tid; idx < FRAG_HALVES; idx += 256) {
    const int split = idx / (FRAG_HALVES / 2);
    const int rem = idx % (FRAG_HALVES / 2);
    const int tile = rem / 512;
    const int i = tile / NK, kt = tile % NK;
    const int lane = (rem % 512) / 16, e = rem % 16;
    const int half = lane >> 4;
    const int m = i * 16 + (lane & 15);
    const int K = kt * 32 + 16 * half + e;
    float a = 0.f;
    if (m < KD && K < KD) {
      float d = 0.f;
      #pragma unroll
      for (int t = 0; t < TT; ++t) d += Dl[m * TT + t] * Dl[K * TT + t];
      a = ((m == K) ? 1.f : 0.f) - d * linv;
    }
    const _Float16 hi = (_Float16)a;
    wsFrag[idx] = split ? (_Float16)(a - (float)hi) : hi;
  }
}

// ---------------------------------------------------------------------------
// Kernel 2: persistent FISTA. One wave = 16 columns, all 40 iterations local.
// ---------------------------------------------------------------------------
__global__ void __launch_bounds__(WAVES * 32, 1)
fista_kernel(const float* __restrict__ x,
             const float* __restrict__ hdr,
             const float* __restrict__ wsD,
             const _Float16* __restrict__ wsFrag,
             float* __restrict__ out) {
  extern __shared__ char smem[];
  _Float16* sFrag = (_Float16*)smem;                  // A hi fragments
  _Float16* sFragLo = sFrag + FRAG_HALVES / 2;        // A lo fragments
  float* sD = (float*)(smem + FRAGB);                 // dictionary, 6464 B

  const int tid = threadIdx.x;
  const int wave = tid >> 5, lane = tid & 31;
  char* wbase = smem + FRAGB + 6464 + (size_t)wave * WAVE_STATE_BYTES;
  _Float16* yfrag = (_Float16*)wbase;                 // [2][NK][32][16] halves
  float* xv = (float*)(wbase + YFRAG_HALVES * 2);     // [NI][2][16][8] floats
  float* dty = xv + DT_FLOATS;                        // same layout

  // Cooperative copy: A fragments + dictionary into LDS (once).
  {
    const uint4* src = (const uint4*)wsFrag;
    uint4* dst = (uint4*)sFrag;
    for (int i = tid; i < FRAG_HALVES / 8; i += blockDim.x) dst[i] = src[i];
    for (int i = tid; i < KD * TT; i += blockDim.x) sD[i] = wsD[i];
  }

  const float linv = hdr[0];
  const float lam = hdr[1];

  const int gcol = blockIdx.x * COLS_PER_WG + wave * 16;
  const int b = gcol / PP, p0 = gcol % PP;   // P % 64 == 0: slab stays in one batch

  // Stage input slab x[b, 0..9, p0..p0+15] into xv (scratch before zeroing).
  for (int i = lane; i < TT * 16; i += 32) {
    const int t = i >> 4, c = i & 15;
    xv[i] = x[((size_t)b * TT + t) * PP + p0 + c];
  }
  __syncthreads();

  // DtY = (1/L) * D^T @ x, written directly in C/D-tile layout.
  for (int idx = lane; idx < NI * 16 * 16; idx += 32) {
    const int k = idx >> 4, c = idx & 15;
    float acc = 0.f;
    if (k < KD) {
      #pragma unroll
      for (int t = 0; t < TT; ++t) acc += sD[k * TT + t] * xv[t * 16 + c];
      acc *= linv;
    }
    const int i = k >> 4, rr = k & 15, hf = rr >> 3, r = rr & 7;
    dty[((i * 2 + hf) * 16 + c) * 8 + r] = acc;
  }
  __syncthreads();
  // Zero x state and y fragments (padding rows stay zero forever).
  for (int i = lane; i < DT_FLOATS; i += 32) xv[i] = 0.f;
  {
    float* yf32 = (float*)yfrag;
    for (int i = lane; i < YFRAG_HALVES / 2; i += 32) yf32[i] = 0.f;
  }
  __syncthreads();

  const int halfid = lane >> 4, colL = lane & 15;
  float tk = 1.f;

  for (int it = 0; it < MAXIT; ++it) {
    // Load all B fragments (already in fragment layout; 24 x ds_load_b128).
    v16h yh[NK], yl[NK];
    #pragma unroll
    for (int kt = 0; kt < NK; ++kt)
      yh[kt] = *(const v16h*)&yfrag[(kt * 32 + lane) * 16];
    #pragma unroll
    for (int kt = 0; kt < NK; ++kt)
      yl[kt] = *(const v16h*)&yfrag[NK * 512 + (kt * 32 + lane) * 16];

    const float tn = 0.5f * (1.f + sqrtf(fmaf(4.f * tk, tk, 1.f)));
    const float mom = (tk - 1.f) / tn;

    for (int i = 0; i < NI; ++i) {
      // Seed accumulator with DtY tile (contiguous 32B per lane).
      const int slot = ((i * 2 + halfid) * 16 + colL) * 8;
      v8f c = *(const v8f*)&dty[slot];

      #pragma unroll
      for (int kt = 0; kt < NK; ++kt) {
        const v16h ah = *(const v16h*)&sFrag[(size_t)(i * NK + kt) * 512 + lane * 16];
        const v16h al = *(const v16h*)&sFragLo[(size_t)(i * NK + kt) * 512 + lane * 16];
        // adjacent uses of the same A operand
        c = __builtin_amdgcn_wmma_f32_16x16x32_f16(false, ah, false, yh[kt], (short)0, c, false, false);
        c = __builtin_amdgcn_wmma_f32_16x16x32_f16(false, ah, false, yl[kt], (short)0, c, false, false);
        c = __builtin_amdgcn_wmma_f32_16x16x32_f16(false, al, false, yh[kt], (short)0, c, false, false);
      }

      // Soft-threshold + Nesterov momentum; pure b128 state updates.
      const v8f xo = *(const v8f*)&xv[slot];
      v8f xn;
      v8h hi8, lo8;
      #pragma unroll
      for (int r = 0; r < 8; ++r) {
        const float s = c[r];
        float m = fabsf(s) - lam;
        m = (m > 0.f) ? m : 0.f;
        const float xnv = copysignf(m, s);
        xn[r] = xnv;
        const float yv = fmaf(mom, xnv - xo[r], xnv);
        const _Float16 h = (_Float16)yv;
        hi8[r] = h;
        lo8[r] = (_Float16)(yv - (float)h);
      }
      *(v8f*)&xv[slot] = xn;
      // y -> B-fragment position: kt = i/2, lane slot colL + 16*(i&1),
      // elements e = 8*half + r (contiguous 16B per split).
      _Float16* ybase = &yfrag[(((i >> 1) * 32 + colL + ((i & 1) << 4)) * 16) + (halfid << 3)];
      *(v8h*)ybase = hi8;
      *(v8h*)(ybase + NK * 512) = lo8;
    }
    tk = tn;
  }

  // Write final x (rows 0..160 only) from tile layout.
  for (int idx = lane; idx < KD * 16; idx += 32) {
    const int k = idx >> 4, c = idx & 15;
    const int i = k >> 4, rr = k & 15, hf = rr >> 3, r = rr & 7;
    out[((size_t)b * KD + k) * PP + p0 + c] = xv[((i * 2 + hf) * 16 + c) * 8 + r];
  }
}

// ---------------------------------------------------------------------------
extern "C" void kernel_launch(void* const* d_in, const int* in_sizes, int n_in,
                              void* d_out, int out_size, void* d_ws, size_t ws_size,
                              hipStream_t stream) {
  (void)in_sizes; (void)n_in; (void)out_size; (void)ws_size;
  const float* x = (const float*)d_in[0];
  const float* Drr = (const float*)d_in[1];
  const float* Dth = (const float*)d_in[2];
  float* out = (float*)d_out;

  float* hdr = (float*)d_ws;
  float* wsD = (float*)((char*)d_ws + WS_D);
  _Float16* wsFrag = (_Float16*)((char*)d_ws + WS_FRAG);

  prep_kernel<<<1, 256, 0, stream>>>(Drr, Dth, hdr, wsD, wsFrag);

  const size_t smem2 = (size_t)FRAGB                  /* A frags f16   */
                     + 6464                           /* dictionary    */
                     + (size_t)WAVES * WAVE_STATE_BYTES; /* y/x/DtY    */
  const int nwg = (BB * PP) / COLS_PER_WG;            // 640 workgroups
  fista_kernel<<<nwg, WAVES * 32, smem2, stream>>>(x, hdr, wsD, wsFrag, out);
}
